// MLPMixer_43860206027125
// MI455X (gfx1250) — matrix operand order repaired and checked by reading, not compiled, not run on hardware
//
#include <hip/hip_runtime.h>
#include <math.h>

#define S_    20
#define U_    512
#define H_    512
#define V_    10000
#define E_    2048   // 4*H
#define K2_   1024   // 2*H
#define VPAD_ 10016  // V padded to multiple of 32

typedef __attribute__((ext_vector_type(16))) _Float16 v16h;
typedef __attribute__((ext_vector_type(8)))  float    v8f;

#define WMMA_F16(a, b, c) \
  __builtin_amdgcn_wmma_f32_16x16x32_f16(false, (a), false, (b), (short)0, (c), false, false)

__device__ __forceinline__ float gelu_t(float x) {
  const float k0 = 0.7978845608028654f; // sqrt(2/pi)
  float x3 = x * x * x;
  return 0.5f * x * (1.0f + tanhf(k0 * (x + 0.044715f * x3)));
}

// ---------------- gathers ----------------
__global__ void k_embed(const int* __restrict__ x, const float* __restrict__ emb,
                        float* __restrict__ xbuf) {
  size_t idx = (size_t)blockIdx.x * blockDim.x + threadIdx.x;   // over S*U*H
  int h = idx % H_;
  size_t su = idx / H_;                                          // s*U+u
  int loc = x[su];
  xbuf[idx] = emb[(size_t)loc * H_ + h];
}

__global__ void k_user(const int* __restrict__ au, const float* __restrict__ uemb,
                       float* __restrict__ pu) {
  int idx = blockIdx.x * blockDim.x + threadIdx.x;               // over U*H
  int h = idx % H_;
  int u = idx / H_;
  pu[idx] = uemb[(size_t)au[u] * H_ + h];
}

// ---------------- layernorm stats: one wave per row of H ----------------
__global__ void k_lnstats(const float* __restrict__ x, float* __restrict__ mean,
                          float* __restrict__ rstd, int nrows) {
  int wid  = threadIdx.x >> 5;
  int lane = threadIdx.x & 31;
  int row  = blockIdx.x * 4 + wid;
  if (row >= nrows) return;
  const float* p = x + (size_t)row * H_;
  float s = 0.f, s2 = 0.f;
  for (int i = lane; i < H_; i += 32) { float v = p[i]; s += v; s2 += v * v; }
  #pragma unroll
  for (int o = 16; o > 0; o >>= 1) {
    s  += __shfl_xor(s,  o, 32);
    s2 += __shfl_xor(s2, o, 32);
  }
  if (lane == 0) {
    float mu  = s / (float)H_;
    float var = s2 / (float)H_ - mu * mu;
    mean[row] = mu;
    rstd[row] = rsqrtf(var + 1e-5f);
  }
}

// ---------------- fused LN1 + token mixing (S=20, per-thread 20x20 matmuls) ----------------
__global__ void k_tokenmix(float* __restrict__ x, const float* __restrict__ mean,
                           const float* __restrict__ rstd,
                           const float* __restrict__ g, const float* __restrict__ b,
                           const float* __restrict__ wt1, const float* __restrict__ wt2) {
  __shared__ float w1[S_ * S_], w2[S_ * S_];
  int t = threadIdx.x;
  for (int i = t; i < S_ * S_; i += blockDim.x) { w1[i] = wt1[i]; w2[i] = wt2[i]; }
  __syncthreads();
  int idx = blockIdx.x * blockDim.x + t;       // over U*H
  int h = idx % H_;
  int u = idx / H_;
  float gv = g[h], bv = b[h];
  float y[S_];
  #pragma unroll
  for (int s = 0; s < S_; s++) {
    float xv = x[(size_t)s * (U_ * H_) + idx];
    int r = s * U_ + u;
    y[s] = (xv - mean[r]) * rstd[r] * gv + bv;
  }
  float t1[S_];
  #pragma unroll
  for (int k = 0; k < S_; k++) {
    float acc = 0.f;
    #pragma unroll
    for (int s = 0; s < S_; s++) acc += y[s] * w1[s * S_ + k];
    t1[k] = gelu_t(acc);
  }
  #pragma unroll
  for (int tt = 0; tt < S_; tt++) {
    float acc = 0.f;
    #pragma unroll
    for (int k = 0; k < S_; k++) acc += t1[k] * w2[k * S_ + tt];
    x[(size_t)tt * (U_ * H_) + idx] += acc;
  }
}

// ---------------- LN apply -> f16 A matrix ----------------
__global__ void k_ln_apply(const float* __restrict__ x, const float* __restrict__ mean,
                           const float* __restrict__ rstd, const float* __restrict__ g,
                           const float* __restrict__ b, _Float16* __restrict__ out, int ldo) {
  size_t idx = (size_t)blockIdx.x * blockDim.x + threadIdx.x;    // over S*U*H
  int h = idx % H_;
  size_t row = idx / H_;
  float v = (x[idx] - mean[row]) * rstd[row] * g[h] + b[h];
  out[row * ldo + h] = (_Float16)v;
}

// ---------------- weight convert+transpose: w[K,N] f32 -> wT[Npad,K] f16 (zero pad) ----------------
__global__ void k_convT(const float* __restrict__ w, _Float16* __restrict__ wT,
                        int K, int N, int Npad) {
  size_t idx = (size_t)blockIdx.x * blockDim.x + threadIdx.x;    // over Npad*K
  int k = idx % K;
  int n = idx / K;
  wT[idx] = (n < N) ? (_Float16)w[(size_t)k * N + n] : (_Float16)0.0f;
}

// ---------------- concat(out_w, p_u) -> f16 [S*U, 2H] ----------------
__global__ void k_concat(const float* __restrict__ x, const float* __restrict__ pu,
                         _Float16* __restrict__ a16) {
  size_t idx = (size_t)blockIdx.x * blockDim.x + threadIdx.x;    // over S*U*H
  int h = idx % H_;
  size_t row = idx / H_;                                          // s*U+u
  int u = (int)(row % U_);
  a16[row * K2_ + h]        = (_Float16)x[idx];
  a16[row * K2_ + H_ + h]   = (_Float16)pu[(size_t)u * H_ + h];
}

// ---------------- WMMA fragment loaders (ISA 7.12.2 layouts, wave32) ----------------
__device__ __forceinline__ v16h load_a_frag(const _Float16* __restrict__ A, int lda,
                                            int m0, int k0, int lane) {
  int r    = lane & 15;
  int koff = (lane >> 4) * 8;                 // lanes 0-15: K 0..7/16..23 ; 16-31: 8..15/24..31
  const _Float16* p = A + (size_t)(m0 + r) * lda + k0 + koff;
  v16h f;
  #pragma unroll
  for (int i = 0; i < 8; i++) f[i] = p[i];
  #pragma unroll
  for (int i = 0; i < 8; i++) f[8 + i] = p[16 + i];
  return f;
}

__device__ __forceinline__ v16h load_b_frag(const _Float16* __restrict__ BT, int ldb,
                                            int n0, int k0, int lane) {
  int c    = lane & 15;
  int koff = (lane >> 4) * 8;                 // lanes 0-15: K 0..7/16..23 ; 16-31: 8..15/24..31
  const _Float16* p = BT + (size_t)(n0 + c) * ldb + k0 + koff;
  v16h f;
  #pragma unroll
  for (int i = 0; i < 8; i++) f[i] = p[i];
  #pragma unroll
  for (int i = 0; i < 8; i++) f[8 + i] = p[16 + i];
  return f;
}

// ---------------- WMMA GEMM: C[M,N] = A[M,K] * B[K,N], B pre-transposed [Npad,K] f16 ----------------
// MODE 0: outH = f16(gelu(acc))        (ldo = N)
// MODE 1: outF += acc                  (residual)
// MODE 2: outF = acc + bias[n], n < N  (final FC)
template <int MODE>
__global__ __launch_bounds__(256) void k_gemm(const _Float16* __restrict__ A, int lda,
                                              const _Float16* __restrict__ BT, int K,
                                              int Npad, int N,
                                              float* __restrict__ outF,
                                              _Float16* __restrict__ outH,
                                              const float* __restrict__ bias, int ldo) {
  const int lane = threadIdx.x & 31;
  const int wid  = threadIdx.x >> 5;
  const int n0   = (blockIdx.x * 8 + wid) * 32;                  // wave-uniform guard
  if (n0 >= Npad) return;
  const int m0 = blockIdx.y * 32;

  v8f c00 = {}, c01 = {}, c10 = {}, c11 = {};
  for (int k0 = 0; k0 < K; k0 += 32) {
    v16h a0 = load_a_frag(A, lda, m0,      k0, lane);
    v16h a1 = load_a_frag(A, lda, m0 + 16, k0, lane);
    v16h b0 = load_b_frag(BT, K, n0,      k0, lane);
    v16h b1 = load_b_frag(BT, K, n0 + 16, k0, lane);
    c00 = WMMA_F16(a0, b0, c00);
    c01 = WMMA_F16(a0, b1, c01);
    c10 = WMMA_F16(a1, b0, c10);
    c11 = WMMA_F16(a1, b1, c11);
  }

  const int ro = (lane >> 4) * 8;   // C layout: VGPR i -> row i (lanes 0-15) / i+8 (lanes 16-31)
  const int cl = lane & 15;
  #pragma unroll
  for (int i = 0; i < 8; i++) {
    int r0 = m0 + ro + i;
    int r1 = r0 + 16;
    int cA = n0 + cl;
    int cB = cA + 16;
    if (MODE == 0) {
      outH[(size_t)r0 * ldo + cA] = (_Float16)gelu_t(c00[i]);
      outH[(size_t)r0 * ldo + cB] = (_Float16)gelu_t(c01[i]);
      outH[(size_t)r1 * ldo + cA] = (_Float16)gelu_t(c10[i]);
      outH[(size_t)r1 * ldo + cB] = (_Float16)gelu_t(c11[i]);
    } else if (MODE == 1) {
      outF[(size_t)r0 * ldo + cA] += c00[i];
      outF[(size_t)r0 * ldo + cB] += c01[i];
      outF[(size_t)r1 * ldo + cA] += c10[i];
      outF[(size_t)r1 * ldo + cB] += c11[i];
    } else {
      if (cA < N) {
        outF[(size_t)r0 * ldo + cA] = c00[i] + bias[cA];
        outF[(size_t)r1 * ldo + cA] = c10[i] + bias[cA];
      }
      if (cB < N) {
        outF[(size_t)r0 * ldo + cB] = c01[i] + bias[cB];
        outF[(size_t)r1 * ldo + cB] = c11[i] + bias[cB];
      }
    }
  }
}

// ---------------- per-u gram matrix sims[i,j,u] = <out[i,u,:], out[j,u,:]> ----------------
#define GCHUNK 512
__global__ void k_gram(const float* __restrict__ out, float* __restrict__ sims) {
  __shared__ float sh[S_][GCHUNK + 4];
  int u = blockIdx.x;
  int t = threadIdx.x;
  // map thread -> lower-tri pair (i >= j), 210 pairs
  int i = 0, j = 0;
  if (t < 210) {
    int p = t;
    while (p >= i + 1) { p -= (i + 1); i++; }
    j = p;
  }
  float acc = 0.f;
  for (int v0 = 0; v0 < V_; v0 += GCHUNK) {
    for (int idx = t; idx < S_ * GCHUNK; idx += 256) {
      int s = idx / GCHUNK, c = idx % GCHUNK;
      int v = v0 + c;
      sh[s][c] = (v < V_) ? out[(size_t)s * (U_ * (size_t)V_) + (size_t)u * V_ + v] : 0.f;
    }
    __syncthreads();
    if (t < 210) {
      #pragma unroll 4
      for (int c = 0; c < GCHUNK; c++) acc += sh[i][c] * sh[j][c];
    }
    __syncthreads();
  }
  if (t < 210) {
    sims[((size_t)i * S_ + j) * U_ + u] = acc;
    sims[((size_t)j * S_ + i) * U_ + u] = acc;
  }
}

// ---------------- causal exp/normalize finisher ----------------
__global__ void k_sdist(const float* __restrict__ sims, float* __restrict__ sd) {
  int idx = blockIdx.x * blockDim.x + threadIdx.x;  // over S*U
  if (idx >= S_ * U_) return;
  int i = idx / U_, u = idx % U_;
  float ni = sqrtf(sims[((size_t)i * S_ + i) * U_ + u]);
  float a[S_];
  float sum = 0.f;
  #pragma unroll
  for (int j = 0; j < S_; j++) {
    float nj  = sqrtf(sims[((size_t)j * S_ + j) * U_ + u]);
    float cs  = sims[((size_t)i * S_ + j) * U_ + u] / fmaxf(ni * nj, 1e-8f);
    float av;
    if (j <= i) { av = expf(fmaxf(cs, 0.f)) + 1e-10f; sum += av; }
    else        { av = 1e-10f; }
    a[j] = av;
  }
  float inv = 1.f / sum;
  #pragma unroll
  for (int j = 0; j < S_; j++) sd[((size_t)i * S_ + j) * U_ + u] = a[j] * inv;
}

// ---------------- host launcher ----------------
extern "C" void kernel_launch(void* const* d_in, const int* in_sizes, int n_in,
                              void* d_out, int out_size, void* d_ws, size_t ws_size,
                              hipStream_t stream) {
  (void)in_sizes; (void)n_in; (void)out_size; (void)ws_size;
  const int*   x        = (const int*)  d_in[0];
  const int*   au       = (const int*)  d_in[2];
  const float* enc_emb  = (const float*)d_in[3];
  const float* user_emb = (const float*)d_in[4];
  const float* blk[2][8];
  for (int b = 0; b < 2; b++)
    for (int i = 0; i < 8; i++) blk[b][i] = (const float*)d_in[5 + b * 8 + i];
  const float* fc_w = (const float*)d_in[21];
  const float* fc_b = (const float*)d_in[22];

  char* ws = (char*)d_ws;
  float*    xbuf = (float*)   (ws + 0);          // 20,971,520  [S*U, H] f32
  float*    pu   = (float*)   (ws + 20971520);   //  1,048,576  [U, H] f32
  float*    mean = (float*)   (ws + 22020096);   //     40,960
  float*    rstd = (float*)   (ws + 22061056);   //     40,960
  _Float16* a16  = (_Float16*)(ws + 22102016);   // 20,971,520  [S*U, up to 2H] f16
  _Float16* h1   = (_Float16*)(ws + 43073536);   // 41,943,040  [S*U, 4H] f16
  _Float16* wT1  = (_Float16*)(ws + 85016576);   //  2,097,152  wc1^T f16
  _Float16* wT2  = (_Float16*)(ws + 87113728);   //  2,097,152  wc2^T f16
  _Float16* fcT  = (_Float16*)(ws + 89210880);   // 20,512,768  fc_w^T padded f16
  float*    sims = (float*)   (ws + 109723648);  //    819,200  [S,S,U] f32

  float* outp  = (float*)d_out;                  // [S,U,V]
  float* sdist = outp + (size_t)S_ * U_ * V_;    // [S,S,U]

  k_embed<<<(S_ * U_ * H_) / 256, 256, 0, stream>>>(x, enc_emb, xbuf);
  k_user<<<(U_ * H_) / 256, 256, 0, stream>>>(au, user_emb, pu);

  for (int b = 0; b < 2; b++) {
    k_lnstats<<<(S_ * U_ + 3) / 4, 128, 0, stream>>>(xbuf, mean, rstd, S_ * U_);
    k_tokenmix<<<(U_ * H_) / 256, 256, 0, stream>>>(xbuf, mean, rstd,
                                                    blk[b][0], blk[b][1], blk[b][2], blk[b][3]);
    k_lnstats<<<(S_ * U_ + 3) / 4, 128, 0, stream>>>(xbuf, mean, rstd, S_ * U_);
    k_ln_apply<<<(S_ * U_ * H_) / 256, 256, 0, stream>>>(xbuf, mean, rstd,
                                                         blk[b][4], blk[b][5], a16, H_);
    k_convT<<<(E_ * H_) / 256, 256, 0, stream>>>(blk[b][6], wT1, H_, E_, E_);   // wc1 [H,4H]
    dim3 g1(E_ / 32 / 8, (S_ * U_) / 32);
    k_gemm<0><<<g1, 256, 0, stream>>>(a16, H_, wT1, H_, E_, E_, nullptr, h1, nullptr, E_);
    k_convT<<<(H_ * E_) / 256, 256, 0, stream>>>(blk[b][7], wT2, E_, H_, H_);   // wc2 [4H,H]
    dim3 g2(H_ / 32 / 8, (S_ * U_) / 32);
    k_gemm<1><<<g2, 256, 0, stream>>>(h1, E_, wT2, E_, H_, H_, xbuf, nullptr, nullptr, H_);
  }

  k_concat<<<(S_ * U_ * H_) / 256, 256, 0, stream>>>(xbuf, pu, a16);
  k_convT<<<(VPAD_ * K2_) / 256, 256, 0, stream>>>(fc_w, fcT, K2_, V_, VPAD_);
  dim3 g3((VPAD_ / 32 + 7) / 8, (S_ * U_) / 32);
  k_gemm<2><<<g3, 256, 0, stream>>>(a16, K2_, fcT, K2_, VPAD_, V_, outp, nullptr, fc_b, V_);

  k_gram<<<U_, 256, 0, stream>>>(outp, sims);
  k_sdist<<<(S_ * U_ + 255) / 256, 256, 0, stream>>>(sims, sdist);
}
